// NLgate_28398323761263
// MI455X (gfx1250) — matrix operand order repaired and checked by reading, not compiled, not run on hardware
//
#include <hip/hip_runtime.h>
#include <hip/hip_bf16.h>
#include <stdint.h>
#include <stddef.h>

// ---------------------------------------------------------------------------
// MI455X / gfx1250 implementation.
// Heavy math on v_wmma_f32_16x16x32_bf16 (wave32); Q-tile staging via
// global_load_async_to_lds_b64 (ASYNCcnt); K/V tile prefetch via
// global_prefetch_b8.
// ---------------------------------------------------------------------------

typedef __bf16 bf16_t;
typedef __attribute__((ext_vector_type(16))) __bf16 v16bf;
typedef __attribute__((ext_vector_type(8)))  float  v8f;

__device__ __forceinline__ bf16_t f2bf(float f) {
  union { float f; uint32_t u; } v; v.f = f;
  uint32_t r = v.u + 0x7FFFu + ((v.u >> 16) & 1u);  // round-to-nearest-even
  uint16_t h = (uint16_t)(r >> 16);
  bf16_t o; __builtin_memcpy(&o, &h, 2); return o;
}

// Load one 16-element bf16 fragment as two 16B chunks (ds_load_b128 x2)
__device__ __forceinline__ v16bf ld_frag(const bf16_t* p0, const bf16_t* p1) {
  union { v16bf v; uint4 q[2]; } u;
  u.q[0] = *(const uint4*)p0;
  u.q[1] = *(const uint4*)p1;
  return u.v;
}

__device__ __forceinline__ v8f wmma_bf16(v16bf a, v16bf b, v8f c) {
  return __builtin_amdgcn_wmma_f32_16x16x32_bf16(
      /*neg_a=*/false, a, /*neg_b=*/false, b,
      /*c_mod=*/(short)0, c, /*reuse_a=*/false, /*reuse_b=*/false);
}

// Async global->LDS copy (8 bytes per lane), tracked by ASYNCcnt.
// LDS destination = low 32 bits of generic shared pointer (LDS aperture).
__device__ __forceinline__ void async_copy_b64(void* lds_dst, const void* gsrc) {
  uint32_t loff = (uint32_t)(uintptr_t)lds_dst;
  uint64_t gaddr = (uint64_t)(uintptr_t)gsrc;
  asm volatile("global_load_async_to_lds_b64 %0, %1, off"
               :: "v"(loff), "v"(gaddr) : "memory");
}
__device__ __forceinline__ void wait_async0() {
  asm volatile("s_wait_asynccnt 0" ::: "memory");
}

// ---------------------------------------------------------------------------
// Kernel 0: elementwise f32 -> bf16 (weights pre-conversion; one pass)
// ---------------------------------------------------------------------------
__global__ void cast_f32_bf16(const float* __restrict__ src,
                              bf16_t* __restrict__ dst, int n) {
  int i = (blockIdx.x * blockDim.x + threadIdx.x) * 4;
  if (i < n) {
    float4 v = *(const float4*)(src + i);
    bf16_t e[4] = {f2bf(v.x), f2bf(v.y), f2bf(v.z), f2bf(v.w)};
    uint2 p; __builtin_memcpy(&p, e, 8);
    *(uint2*)(dst + i) = p;
  }
}

// ---------------------------------------------------------------------------
// Kernel 1: Y = X @ W + b   (X:[M][K] f32, W:[K][N] bf16, Y bf16 [+ f32])
// Block: 256 threads (8 waves). Block tile 128x128, wave tile 32x64.
// ---------------------------------------------------------------------------
__launch_bounds__(256, 2)
__global__ void proj_gemm(const float* __restrict__ X,
                          const bf16_t* __restrict__ Wb,
                          const float* __restrict__ bias,
                          bf16_t* __restrict__ Ybf,
                          float*  __restrict__ Yf,
                          int M, int N, int K) {
  __shared__ bf16_t As[2][128][32];   // [m][k]
  __shared__ bf16_t Bs[2][128][32];   // [n][k] (transposed during load)

  const int tid  = threadIdx.x;
  const int lane = tid & 31;
  const int wave = tid >> 5;
  const int wm   = wave & 3;          // 4 waves along M
  const int wn   = wave >> 2;         // 2 waves along N
  const int m0   = blockIdx.y * 128;
  const int n0   = blockIdx.x * 128;
  const int lr   = lane & 15;
  const int lh   = lane >> 4;

  v8f zero = {};
  v8f acc[2][4];
  #pragma unroll
  for (int i = 0; i < 2; ++i)
    #pragma unroll
    for (int j = 0; j < 4; ++j) acc[i][j] = zero;

  auto loadTiles = [&](int buf, int k0) {
    #pragma unroll
    for (int j = 0; j < 4; ++j) {
      int f = tid + 256 * j;                 // 0..1023 (quad index)
      // A tile: 128 rows x 32 cols f32 (8 quads per row), convert to bf16
      int ar = f >> 3;
      int ac = (f & 7) << 2;
      float4 av = *(const float4*)(X + (size_t)(m0 + ar) * K + k0 + ac);
      As[buf][ar][ac + 0] = f2bf(av.x);
      As[buf][ar][ac + 1] = f2bf(av.y);
      As[buf][ar][ac + 2] = f2bf(av.z);
      As[buf][ar][ac + 3] = f2bf(av.w);
      // B tile: 32 k-rows x 128 cols bf16 (32 quads per row), store transposed
      int bk = f >> 5;
      int bn = (f & 31) << 2;
      uint2 d = *(const uint2*)(Wb + (size_t)(k0 + bk) * N + n0 + bn);
      bf16_t e[4]; __builtin_memcpy(e, &d, 8);
      Bs[buf][bn + 0][bk] = e[0];
      Bs[buf][bn + 1][bk] = e[1];
      Bs[buf][bn + 2][bk] = e[2];
      Bs[buf][bn + 3][bk] = e[3];
    }
  };

  loadTiles(0, 0);
  __syncthreads();
  int cur = 0;
  const int ksteps = K >> 5;
  for (int kt = 0; kt < ksteps; ++kt) {
    if (kt + 1 < ksteps) loadTiles(cur ^ 1, (kt + 1) << 5);
    v16bf a[2];
    #pragma unroll
    for (int mf = 0; mf < 2; ++mf) {
      const bf16_t* ap = &As[cur][wm * 32 + mf * 16 + lr][lh * 8];
      a[mf] = ld_frag(ap, ap + 16);
    }
    #pragma unroll
    for (int nf = 0; nf < 4; ++nf) {
      const bf16_t* bp = &Bs[cur][wn * 64 + nf * 16 + lr][lh * 8];
      v16bf b = ld_frag(bp, bp + 16);
      #pragma unroll
      for (int mf = 0; mf < 2; ++mf)
        acc[mf][nf] = wmma_bf16(a[mf], b, acc[mf][nf]);
    }
    __syncthreads();
    cur ^= 1;
  }

  // Epilogue: + bias, write bf16 (and optionally f32)
  #pragma unroll
  for (int nf = 0; nf < 4; ++nf) {
    int col = n0 + wn * 64 + nf * 16 + lr;
    float bv = bias[col];
    #pragma unroll
    for (int mf = 0; mf < 2; ++mf) {
      int row = m0 + wm * 32 + mf * 16 + lh * 8;
      #pragma unroll
      for (int i = 0; i < 8; ++i) {
        float r = acc[mf][nf][i] + bv;
        Ybf[(size_t)(row + i) * N + col] = f2bf(r);
        if (Yf) Yf[(size_t)(row + i) * N + col] = r;
      }
    }
  }
}

// ---------------------------------------------------------------------------
// Kernel 2: per block: 32 query rows of one batch, full TF=1024 in LDS.
//   S = Qb @ Kin  -> softmax rows in LDS -> Out = P @ Vin + Qf
// LDS: sS 32x1024 f32 (128KB, reused as double-buffered B tiles),
//      sP 32x1024 bf16 (64KB), sA 2x32x32 bf16 (4KB), sRed 32x8 f32 (1KB).
// ---------------------------------------------------------------------------
__launch_bounds__(256, 1)
__global__ void attn_kernel(const bf16_t* __restrict__ Qb,  // [B][1024][1024]
                            const float*  __restrict__ Qf,  // [B][1024][1024]
                            const bf16_t* __restrict__ Kb,  // [B][1024][1024] (C x TF)
                            const bf16_t* __restrict__ Vb,  // [B][1024][1024] (TF x C)
                            float* __restrict__ Out) {      // [B][1024][1024]
  extern __shared__ char smem[];
  float*  sS   = (float*)smem;                          // [32][1024]
  bf16_t* sB   = (bf16_t*)smem;                         // 2 x [1024][32] (aliases sS)
  bf16_t* sP   = (bf16_t*)(smem + 131072);              // [32][1024]
  bf16_t* sA   = (bf16_t*)(smem + 131072 + 65536);      // 2 x [32][32]
  float*  sRed = (float*)(smem + 131072 + 65536 + 4096);// [32][8]

  const int tid  = threadIdx.x;
  const int lane = tid & 31;
  const int wave = tid >> 5;          // wave owns cols [wave*128, wave*128+128)
  const int b    = blockIdx.y;
  const int q0   = blockIdx.x * 32;
  const int lr   = lane & 15;
  const int lh   = lane >> 4;

  const size_t  qbase = ((size_t)b * 1024 + q0) * 1024;
  const bf16_t* kbase = Kb + (size_t)b * 1024 * 1024;
  const bf16_t* vbase = Vb + (size_t)b * 1024 * 1024;

  v8f zero = {};
  v8f acc[2][8];
  #pragma unroll
  for (int i = 0; i < 2; ++i)
    #pragma unroll
    for (int j = 0; j < 8; ++j) acc[i][j] = zero;

  // ---- loaders ----
  auto loadQ = [&](int buf, int k0) {   // 32x32 bf16, async copy, same layout
    int r = tid >> 3;
    int c = (tid & 7) << 2;
    async_copy_b64(&sA[buf * 1024 + r * 32 + c],
                   Qb + qbase + (size_t)r * 1024 + k0 + c);
  };
  auto loadBT = [&](const bf16_t* src, int buf, int k0) { // 32(k) x 1024(n) -> [n][k]
    // prefetch tile two k-steps ahead into L2/WGP$
    if (k0 + 64 < 1024) {
      const bf16_t* pf = src + (size_t)(k0 + 64 + (tid >> 3)) * 1024 + ((tid & 7) << 7);
      __builtin_prefetch(pf, 0, 1);
    }
    for (int j = 0; j < 32; ++j) {
      int f = tid + 256 * j;                 // uint2 index, 8192 total
      int k = f >> 8;                        // 0..31
      int n = (f & 255) << 2;                // 0..1020
      uint2 d = *(const uint2*)(src + (size_t)(k0 + k) * 1024 + n);
      bf16_t e[4]; __builtin_memcpy(e, &d, 8);
      bf16_t* dst = &sB[buf * 32768 + n * 32 + k];
      dst[0] = e[0]; dst[32] = e[1]; dst[64] = e[2]; dst[96] = e[3];
    }
  };

  // ================= Phase 1: S = Q @ Kin =================
  loadQ(0, 0);
  loadBT(kbase, 0, 0);
  wait_async0();
  __syncthreads();
  int cur = 0;
  for (int kt = 0; kt < 32; ++kt) {
    if (kt + 1 < 32) { loadQ(cur ^ 1, (kt + 1) * 32); loadBT(kbase, cur ^ 1, (kt + 1) * 32); }
    v16bf a[2];
    #pragma unroll
    for (int mf = 0; mf < 2; ++mf) {
      const bf16_t* ap = &sA[cur * 1024 + (mf * 16 + lr) * 32 + lh * 8];
      a[mf] = ld_frag(ap, ap + 16);
    }
    #pragma unroll
    for (int nf = 0; nf < 8; ++nf) {
      const bf16_t* bp = &sB[cur * 32768 + (wave * 128 + nf * 16 + lr) * 32 + lh * 8];
      v16bf bb = ld_frag(bp, bp + 16);
      acc[0][nf] = wmma_bf16(a[0], bb, acc[0][nf]);
      acc[1][nf] = wmma_bf16(a[1], bb, acc[1][nf]);
    }
    wait_async0();       // own async Q-copies done before the barrier
    __syncthreads();
    cur ^= 1;
  }

  // spill scores to LDS (overlaps B buffers; all reads done after last sync)
  #pragma unroll
  for (int nf = 0; nf < 8; ++nf) {
    int col = wave * 128 + nf * 16 + lr;
    #pragma unroll
    for (int mf = 0; mf < 2; ++mf)
      #pragma unroll
      for (int i = 0; i < 8; ++i)
        sS[(mf * 16 + lh * 8 + i) * 1024 + col] = acc[mf][nf][i];
  }
  __syncthreads();

  // ================= Phase 2: rowwise softmax =================
  {
    int row = tid >> 3;
    int seg = tid & 7;
    float* srow = sS + row * 1024 + seg * 128;
    float m = -3.402823466e38f;
    for (int i = 0; i < 128; ++i) m = fmaxf(m, srow[i]);
    sRed[row * 8 + seg] = m;
    __syncthreads();
    float rm = sRed[row * 8];
    #pragma unroll
    for (int i = 1; i < 8; ++i) rm = fmaxf(rm, sRed[row * 8 + i]);
    __syncthreads();
    float s = 0.f;
    for (int i = 0; i < 128; ++i) { float p = __expf(srow[i] - rm); srow[i] = p; s += p; }
    sRed[row * 8 + seg] = s;
    __syncthreads();
    float rs = 0.f;
    #pragma unroll
    for (int i = 0; i < 8; ++i) rs += sRed[row * 8 + i];
    float inv = 1.f / rs;
    bf16_t* prow = sP + row * 1024 + seg * 128;
    for (int i = 0; i < 128; ++i) prow[i] = f2bf(srow[i] * inv);
  }
  __syncthreads();

  // ================= Phase 3: Out = P @ Vin + Qf =================
  #pragma unroll
  for (int i = 0; i < 2; ++i)
    #pragma unroll
    for (int j = 0; j < 8; ++j) acc[i][j] = zero;

  loadBT(vbase, 0, 0);
  __syncthreads();
  cur = 0;
  for (int kt = 0; kt < 32; ++kt) {
    if (kt + 1 < 32) loadBT(vbase, cur ^ 1, (kt + 1) * 32);
    const int k0 = kt * 32;
    v16bf a[2];
    #pragma unroll
    for (int mf = 0; mf < 2; ++mf) {
      const bf16_t* ap = &sP[(mf * 16 + lr) * 1024 + k0 + lh * 8];
      a[mf] = ld_frag(ap, ap + 16);
    }
    #pragma unroll
    for (int nf = 0; nf < 8; ++nf) {
      const bf16_t* bp = &sB[cur * 32768 + (wave * 128 + nf * 16 + lr) * 32 + lh * 8];
      v16bf bb = ld_frag(bp, bp + 16);
      acc[0][nf] = wmma_bf16(a[0], bb, acc[0][nf]);
      acc[1][nf] = wmma_bf16(a[1], bb, acc[1][nf]);
    }
    __syncthreads();
    cur ^= 1;
  }

  // Epilogue: residual add + store
  #pragma unroll
  for (int nf = 0; nf < 8; ++nf) {
    int col = wave * 128 + nf * 16 + lr;
    #pragma unroll
    for (int mf = 0; mf < 2; ++mf)
      #pragma unroll
      for (int i = 0; i < 8; ++i) {
        int row = mf * 16 + lh * 8 + i;
        size_t idx = qbase + (size_t)row * 1024 + col;
        Out[idx] = acc[mf][nf][i] + Qf[idx];
      }
  }
}

// ---------------------------------------------------------------------------
// Host-side launcher
// ---------------------------------------------------------------------------
extern "C" void kernel_launch(void* const* d_in, const int* in_sizes, int n_in,
                              void* d_out, int out_size, void* d_ws, size_t ws_size,
                              hipStream_t stream) {
  (void)in_sizes; (void)n_in; (void)out_size; (void)ws_size;
  const float* q  = (const float*)d_in[0];
  const float* k  = (const float*)d_in[1];
  const float* v  = (const float*)d_in[2];
  const float* Wq = (const float*)d_in[3];
  const float* bq = (const float*)d_in[4];
  const float* Wk = (const float*)d_in[5];
  const float* bk = (const float*)d_in[6];
  const float* Wv = (const float*)d_in[7];
  const float* bv = (const float*)d_in[8];
  float* out = (float*)d_out;

  const size_t MB = 1024ull * 1024ull;
  char* ws = (char*)d_ws;
  float*  qf  = (float*)ws;                    // 64 MB: qin f32 (residual)
  bf16_t* qb  = (bf16_t*)(ws + 64 * MB);       // 32 MB
  bf16_t* kb  = (bf16_t*)(ws + 96 * MB);       // 32 MB
  bf16_t* vb  = (bf16_t*)(ws + 128 * MB);      // 32 MB
  bf16_t* wqb = (bf16_t*)(ws + 160 * MB);      // 2 MB
  bf16_t* wkb = (bf16_t*)(ws + 162 * MB);      // 2 MB
  bf16_t* wvb = (bf16_t*)(ws + 164 * MB);      // 2 MB

  const int M = 16384, N = 1024, K = 1024;
  dim3 block(256);

  // weights f32 -> bf16 (once)
  const int wn = 1024 * 1024;
  dim3 cgrid((wn / 4 + 255) / 256);
  cast_f32_bf16<<<cgrid, block, 0, stream>>>(Wq, wqb, wn);
  cast_f32_bf16<<<cgrid, block, 0, stream>>>(Wk, wkb, wn);
  cast_f32_bf16<<<cgrid, block, 0, stream>>>(Wv, wvb, wn);

  dim3 pgrid(N / 128, M / 128);
  proj_gemm<<<pgrid, block, 0, stream>>>(q, wqb, bq, qb, qf, M, N, K);
  proj_gemm<<<pgrid, block, 0, stream>>>(k, wkb, bk, kb, nullptr, M, N, K);
  proj_gemm<<<pgrid, block, 0, stream>>>(v, wvb, bv, vb, nullptr, M, N, K);

  const size_t smem = 131072 + 65536 + 4096 + 1024;  // 201,728 B (<320KB WGP LDS)
  attn_kernel<<<dim3(32, 16), block, smem, stream>>>(qb, qf, kb, vb, out);
}
